// S5Regressor_9354438771047
// MI455X (gfx1250) — matrix operand order repaired and checked by reading, not compiled, not run on hardware
//
#include <hip/hip_runtime.h>
#include <math.h>

// ---------------------------------------------------------------------------
// S5 regressor forward for MI455X (gfx1250): bf16 WMMA GEMMs + chunked scan.
// ---------------------------------------------------------------------------

#define DMODEL 1048
#define DPAD   1088            // 64*17, multiple of 32 for K, 64 for N tiles
#define LSEQ   4096
#define BSZ    4
#define MROWS  (BSZ * LSEQ)    // 16384
#define NCHUNK 16
#define CLEN   (LSEQ / NCHUNK) // 256
#define NLAYER 3

typedef __attribute__((ext_vector_type(16))) __bf16 v16bf;
typedef __attribute__((ext_vector_type(8)))  float  v8f;

// ---------------- WMMA fragment helpers (wave32, 16x16x32 bf16) -------------

union FragU { v16bf v; uint4 q[2]; };

// A: [M,K] row-major. Lane (half=lane>>4, m=lane&15):
//   elems 0..7  -> K = half*8 + 0..7
//   elems 8..15 -> K = 16 + half*8 + 0..7
__device__ __forceinline__ v16bf load_a_frag(const __bf16* __restrict__ A,
                                             int ld, int m0, int k0) {
  int lane = threadIdx.x & 31;
  int half = lane >> 4;
  const __bf16* p = A + (size_t)(m0 + (lane & 15)) * ld + k0 + half * 8;
  FragU u;
  u.q[0] = *(const uint4*)p;
  u.q[1] = *(const uint4*)(p + 16);
  return u.v;
}

// B as W[N,K] row-major (we compute X @ W^T). Lane (khalf=lane>>4, n=lane&15)
// holds W[n][khalf*8 .. khalf*8+7] and W[n][16+khalf*8 .. 16+khalf*8+7].
__device__ __forceinline__ v16bf load_b_frag(const __bf16* __restrict__ W,
                                             int ld, int n0, int k0) {
  int lane = threadIdx.x & 31;
  int khalf = lane >> 4;
  const __bf16* p = W + (size_t)(n0 + (lane & 15)) * ld + k0 + khalf * 8;
  FragU u;
  u.q[0] = *(const uint4*)p;
  u.q[1] = *(const uint4*)(p + 16);
  return u.v;
}

__device__ __forceinline__ v8f wmma_bf16(v16bf a, v16bf b, v8f c) {
  return __builtin_amdgcn_wmma_f32_16x16x32_bf16(false, a, false, b,
                                                 (short)0, c, false, false);
}

// C/D 16x16 f32: lane (mh=lane>>4, n=lane&15), VGPR r -> (M = mh*8+r, N = n)
__device__ __forceinline__ void store_c_frag(float* __restrict__ C, int ldc,
                                             int m0, int n0, v8f c) {
  int lane = threadIdx.x & 31;
  int n = n0 + (lane & 15);
  int mh = (lane >> 4) * 8;
  size_t base = (size_t)(m0 + mh) * ldc + n;
#pragma unroll
  for (int r = 0; r < 8; ++r) C[base + (size_t)r * ldc] = c[r];
}

__device__ __forceinline__ float gelu_f(float x) {
  return 0.5f * x * (1.0f + erff(x * 0.70710678118654752f));
}

// ---------------- GEMM kernels: block = 8 waves -> 128x64 tile --------------
// wave (wm = wave>>1 in [0,4), wn = wave&1 in [0,2)) computes 32x32 (2x2 frags)

// out0 = A @ B0^T ; out1 = A @ B1^T   (GEMM1: Bu_re / Bu_im, shared A loads)
__global__ __launch_bounds__(256) void gemm_ab2_kernel(
    const __bf16* __restrict__ A, const __bf16* __restrict__ B0,
    const __bf16* __restrict__ B1, float* __restrict__ C0,
    float* __restrict__ C1) {
  int wave = threadIdx.x >> 5;
  int m0 = blockIdx.x * 128 + (wave >> 1) * 32;
  int n0 = blockIdx.y * 64 + (wave & 1) * 32;
  v8f c[2][2] = {};
  v8f d[2][2] = {};
  for (int k = 0; k < DPAD; k += 32) {
    v16bf a0 = load_a_frag(A, DPAD, m0, k);
    v16bf a1 = load_a_frag(A, DPAD, m0 + 16, k);
    v16bf b00 = load_b_frag(B0, DPAD, n0, k);
    v16bf b01 = load_b_frag(B0, DPAD, n0 + 16, k);
    v16bf b10 = load_b_frag(B1, DPAD, n0, k);
    v16bf b11 = load_b_frag(B1, DPAD, n0 + 16, k);
    c[0][0] = wmma_bf16(a0, b00, c[0][0]);
    c[0][1] = wmma_bf16(a0, b01, c[0][1]);
    c[1][0] = wmma_bf16(a1, b00, c[1][0]);
    c[1][1] = wmma_bf16(a1, b01, c[1][1]);
    d[0][0] = wmma_bf16(a0, b10, d[0][0]);
    d[0][1] = wmma_bf16(a0, b11, d[0][1]);
    d[1][0] = wmma_bf16(a1, b10, d[1][0]);
    d[1][1] = wmma_bf16(a1, b11, d[1][1]);
  }
#pragma unroll
  for (int mi = 0; mi < 2; ++mi)
#pragma unroll
    for (int ni = 0; ni < 2; ++ni) {
      store_c_frag(C0, DPAD, m0 + mi * 16, n0 + ni * 16, c[mi][ni]);
      store_c_frag(C1, DPAD, m0 + mi * 16, n0 + ni * 16, d[mi][ni]);
    }
}

// out = A0 @ B0^T + A1 @ B1^T   (GEMM2: xs_re@C_re^T + xs_im@(-C_im)^T)
__global__ __launch_bounds__(256) void gemm_a2b2_kernel(
    const __bf16* __restrict__ A0, const __bf16* __restrict__ B0,
    const __bf16* __restrict__ A1, const __bf16* __restrict__ B1,
    float* __restrict__ C) {
  int wave = threadIdx.x >> 5;
  int m0 = blockIdx.x * 128 + (wave >> 1) * 32;
  int n0 = blockIdx.y * 64 + (wave & 1) * 32;
  v8f c[2][2] = {};
  for (int k = 0; k < DPAD; k += 32) {
    v16bf a00 = load_a_frag(A0, DPAD, m0, k);
    v16bf a01 = load_a_frag(A0, DPAD, m0 + 16, k);
    v16bf a10 = load_a_frag(A1, DPAD, m0, k);
    v16bf a11 = load_a_frag(A1, DPAD, m0 + 16, k);
    v16bf b00 = load_b_frag(B0, DPAD, n0, k);
    v16bf b01 = load_b_frag(B0, DPAD, n0 + 16, k);
    v16bf b10 = load_b_frag(B1, DPAD, n0, k);
    v16bf b11 = load_b_frag(B1, DPAD, n0 + 16, k);
    c[0][0] = wmma_bf16(a00, b00, c[0][0]);
    c[0][1] = wmma_bf16(a00, b01, c[0][1]);
    c[1][0] = wmma_bf16(a01, b00, c[1][0]);
    c[1][1] = wmma_bf16(a01, b01, c[1][1]);
    c[0][0] = wmma_bf16(a10, b10, c[0][0]);
    c[0][1] = wmma_bf16(a10, b11, c[0][1]);
    c[1][0] = wmma_bf16(a11, b10, c[1][0]);
    c[1][1] = wmma_bf16(a11, b11, c[1][1]);
  }
#pragma unroll
  for (int mi = 0; mi < 2; ++mi)
#pragma unroll
    for (int ni = 0; ni < 2; ++ni)
      store_c_frag(C, DPAD, m0 + mi * 16, n0 + ni * 16, c[mi][ni]);
}

// GEGLU: accA = A@Wa^T, accG = A@Wg^T, out(bf16) = accA * gelu(accG)
__global__ __launch_bounds__(256) void gemm_geglu_kernel(
    const __bf16* __restrict__ A, const __bf16* __restrict__ Wa,
    const __bf16* __restrict__ Wg, __bf16* __restrict__ Out) {
  int wave = threadIdx.x >> 5;
  int m0 = blockIdx.x * 128 + (wave >> 1) * 32;
  int n0 = blockIdx.y * 64 + (wave & 1) * 32;
  v8f ca[2][2] = {};
  v8f cg[2][2] = {};
  for (int k = 0; k < DPAD; k += 32) {
    v16bf a0 = load_a_frag(A, DPAD, m0, k);
    v16bf a1 = load_a_frag(A, DPAD, m0 + 16, k);
    v16bf ba0 = load_b_frag(Wa, DPAD, n0, k);
    v16bf ba1 = load_b_frag(Wa, DPAD, n0 + 16, k);
    v16bf bg0 = load_b_frag(Wg, DPAD, n0, k);
    v16bf bg1 = load_b_frag(Wg, DPAD, n0 + 16, k);
    ca[0][0] = wmma_bf16(a0, ba0, ca[0][0]);
    ca[0][1] = wmma_bf16(a0, ba1, ca[0][1]);
    ca[1][0] = wmma_bf16(a1, ba0, ca[1][0]);
    ca[1][1] = wmma_bf16(a1, ba1, ca[1][1]);
    cg[0][0] = wmma_bf16(a0, bg0, cg[0][0]);
    cg[0][1] = wmma_bf16(a0, bg1, cg[0][1]);
    cg[1][0] = wmma_bf16(a1, bg0, cg[1][0]);
    cg[1][1] = wmma_bf16(a1, bg1, cg[1][1]);
  }
  int lane = threadIdx.x & 31;
  int n = lane & 15;
  int mh = (lane >> 4) * 8;
#pragma unroll
  for (int mi = 0; mi < 2; ++mi)
#pragma unroll
    for (int ni = 0; ni < 2; ++ni) {
      size_t base = (size_t)(m0 + mi * 16 + mh) * DPAD + n0 + ni * 16 + n;
#pragma unroll
      for (int r = 0; r < 8; ++r)
        Out[base + (size_t)r * DPAD] =
            (__bf16)(ca[mi][ni][r] * gelu_f(cg[mi][ni][r]));
    }
}

// Residual: h = h + (A @ Wd^T) + fy
__global__ __launch_bounds__(256) void gemm_res_kernel(
    const __bf16* __restrict__ A, const __bf16* __restrict__ Wd,
    const float* __restrict__ Fy, float* __restrict__ H) {
  int wave = threadIdx.x >> 5;
  int m0 = blockIdx.x * 128 + (wave >> 1) * 32;
  int n0 = blockIdx.y * 64 + (wave & 1) * 32;
  v8f c[2][2] = {};
  for (int k = 0; k < DPAD; k += 32) {
    v16bf a0 = load_a_frag(A, DPAD, m0, k);
    v16bf a1 = load_a_frag(A, DPAD, m0 + 16, k);
    v16bf b0 = load_b_frag(Wd, DPAD, n0, k);
    v16bf b1 = load_b_frag(Wd, DPAD, n0 + 16, k);
    c[0][0] = wmma_bf16(a0, b0, c[0][0]);
    c[0][1] = wmma_bf16(a0, b1, c[0][1]);
    c[1][0] = wmma_bf16(a1, b0, c[1][0]);
    c[1][1] = wmma_bf16(a1, b1, c[1][1]);
  }
  int lane = threadIdx.x & 31;
  int n = lane & 15;
  int mh = (lane >> 4) * 8;
#pragma unroll
  for (int mi = 0; mi < 2; ++mi)
#pragma unroll
    for (int ni = 0; ni < 2; ++ni) {
      size_t base = (size_t)(m0 + mi * 16 + mh) * DPAD + n0 + ni * 16 + n;
#pragma unroll
      for (int r = 0; r < 8; ++r) {
        size_t idx = base + (size_t)r * DPAD;
        H[idx] = H[idx] + c[mi][ni][r] + Fy[idx];
      }
    }
}

// ---------------- Elementwise / normalization kernels -----------------------

__device__ __forceinline__ float block_reduce(float val, float* red) {
  int tid = threadIdx.x;
  red[tid] = val;
  __syncthreads();
  for (int s = 128; s > 0; s >>= 1) {
    if (tid < s) red[tid] += red[tid + s];
    __syncthreads();
  }
  float r = red[0];
  __syncthreads();
  return r;
}

__global__ __launch_bounds__(256) void encode_kernel(
    const float* __restrict__ x, const float* __restrict__ ew,
    const float* __restrict__ eb, float* __restrict__ h) {
  int m = blockIdx.x;
  float xv = x[m];
  float* row = h + (size_t)m * DPAD;
  for (int d = threadIdx.x; d < DPAD; d += 256)
    row[d] = (d < DMODEL) ? xv * ew[d] + eb[d] : 0.0f;
}

// fused: z = LN(h, nw, nb); fx = LN(z, aw, ab); emit fx f32 + zero-padded bf16
__global__ __launch_bounds__(256) void ln2_kernel(
    const float* __restrict__ h, const float* __restrict__ nw,
    const float* __restrict__ nb, const float* __restrict__ aw,
    const float* __restrict__ ab, float* __restrict__ fx,
    __bf16* __restrict__ fxb) {
  __shared__ float red[256];
  int m = blockIdx.x;
  int tid = threadIdx.x;
  const float* row = h + (size_t)m * DPAD;
  float v[5];
  float s1 = 0.f, s2 = 0.f;
#pragma unroll
  for (int it = 0; it < 5; ++it) {
    int d = tid + it * 256;
    float xv = (d < DMODEL) ? row[d] : 0.0f;
    v[it] = xv;
    s1 += xv;
    s2 += xv * xv;
  }
  s1 = block_reduce(s1, red);
  s2 = block_reduce(s2, red);
  const float invD = 1.0f / DMODEL;
  float mu = s1 * invD;
  float rs = rsqrtf(s2 * invD - mu * mu + 1e-5f);
  float t1 = 0.f, t2 = 0.f;
#pragma unroll
  for (int it = 0; it < 5; ++it) {
    int d = tid + it * 256;
    float z = 0.f;
    if (d < DMODEL) z = (v[it] - mu) * rs * nw[d] + nb[d];
    v[it] = z;
    t1 += z;
    t2 += z * z;
  }
  t1 = block_reduce(t1, red);
  t2 = block_reduce(t2, red);
  float mu2 = t1 * invD;
  float rs2 = rsqrtf(t2 * invD - mu2 * mu2 + 1e-5f);
  float* frow = fx + (size_t)m * DPAD;
  __bf16* brow = fxb + (size_t)m * DPAD;
#pragma unroll
  for (int it = 0; it < 5; ++it) {
    int d = tid + it * 256;
    if (d < DPAD) {
      float o = (d < DMODEL) ? (v[it] - mu2) * rs2 * aw[d] + ab[d] : 0.0f;
      frow[d] = o;
      brow[d] = (__bf16)o;
    }
  }
}

// fused: s = s_pre + Dvec*fx; z = gelu(s) + fx; fy = LN(z, fw, fb)
__global__ __launch_bounds__(256) void ln_post_kernel(
    const float* __restrict__ sp, const float* __restrict__ fx,
    const float* __restrict__ Dv, const float* __restrict__ fw,
    const float* __restrict__ fb, float* __restrict__ fy,
    __bf16* __restrict__ fyb) {
  __shared__ float red[256];
  int m = blockIdx.x;
  int tid = threadIdx.x;
  const float* srow = sp + (size_t)m * DPAD;
  const float* xrow = fx + (size_t)m * DPAD;
  float v[5];
  float s1 = 0.f, s2 = 0.f;
#pragma unroll
  for (int it = 0; it < 5; ++it) {
    int d = tid + it * 256;
    float z = 0.f;
    if (d < DMODEL) {
      float u = xrow[d];
      float t = srow[d] + Dv[d] * u;
      z = gelu_f(t) + u;
    }
    v[it] = z;
    s1 += z;
    s2 += z * z;
  }
  s1 = block_reduce(s1, red);
  s2 = block_reduce(s2, red);
  const float invD = 1.0f / DMODEL;
  float mu = s1 * invD;
  float rs = rsqrtf(s2 * invD - mu * mu + 1e-5f);
  float* frow = fy + (size_t)m * DPAD;
  __bf16* brow = fyb + (size_t)m * DPAD;
#pragma unroll
  for (int it = 0; it < 5; ++it) {
    int d = tid + it * 256;
    if (d < DPAD) {
      float o = (d < DMODEL) ? (v[it] - mu) * rs * fw[d] + fb[d] : 0.0f;
      frow[d] = o;
      brow[d] = (__bf16)o;
    }
  }
}

// ---------------- Discretization & weight conversion ------------------------

__global__ __launch_bounds__(256) void discretize_kernel(
    const float* __restrict__ lam_re, const float* __restrict__ lam_im,
    const float* __restrict__ lstep, const float* __restrict__ B_re,
    const float* __restrict__ B_im, __bf16* __restrict__ bb_re,
    __bf16* __restrict__ bb_im, float* __restrict__ ab_re,
    float* __restrict__ ab_im) {
  int p = blockIdx.x;  // [0, DPAD)
  float cr = 0.f, ci = 0.f, Ar = 0.f, Ai = 0.f;
  if (p < DMODEL) {
    float lr = lam_re[p], li = lam_im[p];
    float st = expf(lstep[p]);
    float er = expf(lr * st);
    Ar = er * cosf(li * st);
    Ai = er * sinf(li * st);
    float nr = Ar - 1.0f, ni = Ai;
    float inv = 1.0f / (lr * lr + li * li);
    cr = (nr * lr + ni * li) * inv;   // (Abar-1)/Lambda
    ci = (ni * lr - nr * li) * inv;
  }
  if (threadIdx.x == 0) {
    ab_re[p] = Ar;
    ab_im[p] = Ai;
  }
  for (int d = threadIdx.x; d < DPAD; d += 256) {
    float br = 0.f, bi = 0.f;
    if (p < DMODEL && d < DMODEL) {
      float xr = B_re[(size_t)p * DMODEL + d];
      float xi = B_im[(size_t)p * DMODEL + d];
      br = cr * xr - ci * xi;
      bi = cr * xi + ci * xr;
    }
    bb_re[(size_t)p * DPAD + d] = (__bf16)br;
    bb_im[(size_t)p * DPAD + d] = (__bf16)bi;
  }
}

// zero-padded f32 -> bf16 conversion (rows R x cols C into DPAD x DPAD)
__global__ __launch_bounds__(256) void convert_pad_kernel(
    const float* __restrict__ src, __bf16* __restrict__ dst, int R, int C,
    float scale) {
  int r = blockIdx.x;  // [0, DPAD)
  for (int c = threadIdx.x; c < DPAD; c += 256) {
    float v = (r < R && c < C) ? src[(size_t)r * C + c] * scale : 0.0f;
    dst[(size_t)r * DPAD + c] = (__bf16)v;
  }
}

// ---------------- Chunked diagonal-complex scan (lanes across p) ------------

__global__ __launch_bounds__(256) void scan_pass1(
    const float* __restrict__ bu_re, const float* __restrict__ bu_im,
    const float* __restrict__ ab_re, const float* __restrict__ ab_im,
    float* __restrict__ s_re, float* __restrict__ s_im) {
  int p = blockIdx.x * 256 + threadIdx.x;
  if (p >= DPAD) return;
  int c = blockIdx.y, b = blockIdx.z;
  float Ar = ab_re[p], Ai = ab_im[p];
  size_t base = ((size_t)b * LSEQ + (size_t)c * CLEN) * DPAD + p;
  float xr = 0.f, xi = 0.f;
  for (int j = 0; j < CLEN; ++j) {
    float br = bu_re[base + (size_t)j * DPAD];
    float bi = bu_im[base + (size_t)j * DPAD];
    float nr = Ar * xr - Ai * xi + br;
    float ni = Ar * xi + Ai * xr + bi;
    xr = nr;
    xi = ni;
  }
  size_t si = ((size_t)b * NCHUNK + c) * DPAD + p;
  s_re[si] = xr;
  s_im[si] = xi;
}

__global__ __launch_bounds__(256) void scan_pass2(
    const float* __restrict__ ab_re, const float* __restrict__ ab_im,
    float* __restrict__ s_re, float* __restrict__ s_im) {
  int p = blockIdx.x * 256 + threadIdx.x;
  if (p >= DPAD) return;
  int b = blockIdx.z;
  float Ar = ab_re[p], Ai = ab_im[p];
#pragma unroll
  for (int t = 0; t < 8; ++t) {  // A^(2^8) = Abar^CLEN
    float nr = Ar * Ar - Ai * Ai;
    Ai = 2.0f * Ar * Ai;
    Ar = nr;
  }
  float cr = 0.f, ci = 0.f;
  for (int c = 0; c < NCHUNK; ++c) {
    size_t si = ((size_t)b * NCHUNK + c) * DPAD + p;
    float tr = s_re[si], ti = s_im[si];
    s_re[si] = cr;  // overwrite with exclusive carry
    s_im[si] = ci;
    float nr = Ar * cr - Ai * ci + tr;
    float ni = Ar * ci + Ai * cr + ti;
    cr = nr;
    ci = ni;
  }
}

__global__ __launch_bounds__(256) void scan_pass3(
    const float* __restrict__ bu_re, const float* __restrict__ bu_im,
    const float* __restrict__ ab_re, const float* __restrict__ ab_im,
    const float* __restrict__ s_re, const float* __restrict__ s_im,
    __bf16* __restrict__ xs_re, __bf16* __restrict__ xs_im) {
  int p = blockIdx.x * 256 + threadIdx.x;
  if (p >= DPAD) return;
  int c = blockIdx.y, b = blockIdx.z;
  float Ar = ab_re[p], Ai = ab_im[p];
  size_t si = ((size_t)b * NCHUNK + c) * DPAD + p;
  float xr = s_re[si], xi = s_im[si];
  size_t base = ((size_t)b * LSEQ + (size_t)c * CLEN) * DPAD + p;
  for (int j = 0; j < CLEN; ++j) {
    size_t idx = base + (size_t)j * DPAD;
    float br = bu_re[idx];
    float bi = bu_im[idx];
    float nr = Ar * xr - Ai * xi + br;
    float ni = Ar * xi + Ai * xr + bi;
    xr = nr;
    xi = ni;
    xs_re[idx] = (__bf16)xr;
    xs_im[idx] = (__bf16)xi;
  }
}

// ---------------- Pooling + head --------------------------------------------

__global__ __launch_bounds__(256) void pool_kernel(
    const float* __restrict__ h, const float* __restrict__ hw,
    float* __restrict__ partials) {
  __shared__ float red[256];
  int b = blockIdx.x, c = blockIdx.y;  // 32 chunks of 128 rows
  const int LC = LSEQ / 32;
  size_t base = ((size_t)b * LSEQ + (size_t)c * LC) * DPAD;
  float acc = 0.f;
  const int tot = LC * DMODEL;
  for (int idx = threadIdx.x; idx < tot; idx += 256) {
    int l = idx / DMODEL;
    int d = idx - l * DMODEL;
    acc += h[base + (size_t)l * DPAD + d] * hw[d];
  }
  acc = block_reduce(acc, red);
  if (threadIdx.x == 0) partials[b * 32 + c] = acc;
}

__global__ void head_kernel(const float* __restrict__ partials,
                            const float* __restrict__ hb,
                            float* __restrict__ out) {
  int b = threadIdx.x;
  if (b < BSZ) {
    float s = 0.f;
    for (int c = 0; c < 32; ++c) s += partials[b * 32 + c];
    out[b] = s * (1.0f / LSEQ) + hb[0];
  }
}

// ---------------- Orchestration ---------------------------------------------

extern "C" void kernel_launch(void* const* d_in, const int* in_sizes, int n_in,
                              void* d_out, int out_size, void* d_ws,
                              size_t ws_size, hipStream_t stream) {
  (void)in_sizes; (void)n_in; (void)out_size; (void)ws_size;
  const int D = DMODEL, P = DMODEL;
  const float* x      = (const float*)d_in[0];
  const float* enc_w  = (const float*)d_in[1];
  const float* enc_b  = (const float*)d_in[2];
  const float* norm_w = (const float*)d_in[3];
  const float* norm_b = (const float*)d_in[4];
  const float* anw    = (const float*)d_in[5];
  const float* anb    = (const float*)d_in[6];
  const float* fnw    = (const float*)d_in[7];
  const float* fnb    = (const float*)d_in[8];
  const float* lam_re = (const float*)d_in[9];
  const float* lam_im = (const float*)d_in[10];
  const float* lstep  = (const float*)d_in[11];
  const float* B_re   = (const float*)d_in[12];
  const float* B_im   = (const float*)d_in[13];
  const float* C_re   = (const float*)d_in[14];
  const float* C_im   = (const float*)d_in[15];
  const float* Dvec   = (const float*)d_in[16];
  const float* Wenc   = (const float*)d_in[17];
  const float* Wdec   = (const float*)d_in[18];
  const float* head_w = (const float*)d_in[19];
  const float* head_b = (const float*)d_in[20];
  float* out = (float*)d_out;

  // ---- workspace layout (lifetime-aliased) ----
  char* w = (char*)d_ws;
  const size_t f32m  = (size_t)MROWS * DPAD * sizeof(float);
  const size_t bf16m = (size_t)MROWS * DPAD * sizeof(__bf16);
  const size_t wmat  = (size_t)DPAD * DPAD * sizeof(__bf16);
  float*  h     = (float*)(w);
  float*  fx    = (float*)(w + f32m);
  float*  bu_re = (float*)(w + 2 * f32m);            // alias: s_pre
  float*  bu_im = (float*)(w + 3 * f32m);            // alias: fy
  __bf16* fx_b  = (__bf16*)(w + 4 * f32m);           // alias: fy_b
  __bf16* xs_re = (__bf16*)(w + 4 * f32m + bf16m);   // alias: geglu
  __bf16* xs_im = (__bf16*)(w + 4 * f32m + 2 * bf16m);
  char* wp = w + 4 * f32m + 3 * bf16m;
  __bf16* w_bbr = (__bf16*)wp; wp += wmat;
  __bf16* w_bbi = (__bf16*)wp; wp += wmat;
  __bf16* w_cre = (__bf16*)wp; wp += wmat;
  __bf16* w_cim = (__bf16*)wp; wp += wmat;   // holds -C_im
  __bf16* w_wa  = (__bf16*)wp; wp += wmat;
  __bf16* w_wg  = (__bf16*)wp; wp += wmat;
  __bf16* w_wd  = (__bf16*)wp; wp += wmat;
  float* ab_re = (float*)wp; wp += DPAD * sizeof(float);
  float* ab_im = (float*)wp; wp += DPAD * sizeof(float);
  float* s_re  = (float*)wp; wp += (size_t)BSZ * NCHUNK * DPAD * sizeof(float);
  float* s_im  = (float*)wp; wp += (size_t)BSZ * NCHUNK * DPAD * sizeof(float);
  float* partials = (float*)wp;

  float*  s_pre = bu_re;
  float*  fy    = bu_im;
  __bf16* fy_b  = fx_b;
  __bf16* geglu = xs_re;

  dim3 ggrid(MROWS / 128, DPAD / 64, 1);
  dim3 sg1((DPAD + 255) / 256, NCHUNK, BSZ);
  dim3 sg2((DPAD + 255) / 256, 1, BSZ);

  encode_kernel<<<MROWS, 256, 0, stream>>>(x, enc_w, enc_b, h);

  for (int i = 0; i < NLAYER; ++i) {
    ln2_kernel<<<MROWS, 256, 0, stream>>>(h, norm_w + i * D, norm_b + i * D,
                                          anw + i * D, anb + i * D, fx, fx_b);
    discretize_kernel<<<DPAD, 256, 0, stream>>>(
        lam_re + i * P, lam_im + i * P, lstep + i * P,
        B_re + (size_t)i * P * D, B_im + (size_t)i * P * D,
        w_bbr, w_bbi, ab_re, ab_im);
    convert_pad_kernel<<<DPAD, 256, 0, stream>>>(C_re + (size_t)i * D * P,
                                                 w_cre, D, P, 1.0f);
    convert_pad_kernel<<<DPAD, 256, 0, stream>>>(C_im + (size_t)i * D * P,
                                                 w_cim, D, P, -1.0f);
    convert_pad_kernel<<<DPAD, 256, 0, stream>>>(Wenc + (size_t)i * 2 * D * D,
                                                 w_wa, D, D, 1.0f);
    convert_pad_kernel<<<DPAD, 256, 0, stream>>>(
        Wenc + (size_t)i * 2 * D * D + (size_t)D * D, w_wg, D, D, 1.0f);
    convert_pad_kernel<<<DPAD, 256, 0, stream>>>(Wdec + (size_t)i * D * D,
                                                 w_wd, D, D, 1.0f);

    // Bu = fx @ Bbar^T (re & im share the A operand)
    gemm_ab2_kernel<<<ggrid, 256, 0, stream>>>(fx_b, w_bbr, w_bbi, bu_re,
                                               bu_im);
    // chunked diagonal-complex scan (coalesced over p)
    scan_pass1<<<sg1, 256, 0, stream>>>(bu_re, bu_im, ab_re, ab_im, s_re, s_im);
    scan_pass2<<<sg2, 256, 0, stream>>>(ab_re, ab_im, s_re, s_im);
    scan_pass3<<<sg1, 256, 0, stream>>>(bu_re, bu_im, ab_re, ab_im, s_re, s_im,
                                        xs_re, xs_im);
    // y = xs_re @ C_re^T + xs_im @ (-C_im)^T  (overwrites Bu_re as s_pre)
    gemm_a2b2_kernel<<<ggrid, 256, 0, stream>>>(xs_re, w_cre, xs_im, w_cim,
                                                s_pre);
    // z = gelu(y + Dvec*fx) + fx ; fy = LN(z)
    ln_post_kernel<<<MROWS, 256, 0, stream>>>(s_pre, fx, Dvec + i * D,
                                              fnw + i * D, fnb + i * D, fy,
                                              fy_b);
    // geglu = (fy@Wa^T) * gelu(fy@Wg^T)
    gemm_geglu_kernel<<<ggrid, 256, 0, stream>>>(fy_b, w_wa, w_wg, geglu);
    // h = h + geglu@Wdec^T + fy
    gemm_res_kernel<<<ggrid, 256, 0, stream>>>(geglu, w_wd, fy, h);
  }

  pool_kernel<<<dim3(BSZ, 32), 256, 0, stream>>>(h, head_w, partials);
  head_kernel<<<1, 64, 0, stream>>>(partials, head_b, out);
}
